// SAGENet_45586782879799
// MI455X (gfx1250) — matrix operand order repaired and checked by reading, not compiled, not run on hardware
//
#include <hip/hip_runtime.h>
#include <hip/hip_bf16.h>

// ---------------------------------------------------------------------------
// GraphSAGE (2-layer) for MI455X / gfx1250.
//
// Pipeline (all buffers in d_ws, all kernels on `stream`):
//   1. transpose X [T=8,N,32] -> XT [N, T*C=256]      (node-major, 1KB rows)
//   2. degree count (global_atomic_add_f32)
//   3. layer-1 edge aggregation: wave32 per edge, coalesced gather + atomics
//   4. pack h1 = concat(x, agg/deg) as f16 [M=N*T, 64]
//   5. GEMM1 via v_wmma_f32_16x16x32_f16 (K=64 -> 2 WMMA steps), fused
//      bias + leaky-ReLU; writes y1 (f32, for layer-2 aggregation) and the
//      first 16 f16 columns of h2
//   6. layer-2 edge aggregation over y1 [M,16] (128 feats/node)
//   7. pack agg2 into h2 cols 16..31 (f16)
//   8. GEMM2 via v_wmma_f32_16x16x32_f16 (K=32 -> 1 WMMA step, 4 B-tiles),
//      fused bias + leaky-ReLU + transpose back to output [T, N, 64]
// ---------------------------------------------------------------------------

typedef __attribute__((ext_vector_type(16))) _Float16 v16h;
typedef __attribute__((ext_vector_type(8)))  _Float16 h8;
typedef __attribute__((ext_vector_type(8)))  float    v8f;

#define LEAKY(v) ((v) > 0.0f ? (v) : 0.01f * (v))

// ---------------- 1. transpose X [T,N,32] -> XT [N,256] --------------------
__global__ __launch_bounds__(256) void transpose_x(
    const float* __restrict__ X, float* __restrict__ XT, int Nn) {
  int tid = blockIdx.x * blockDim.x + threadIdx.x;
  if (tid >= Nn * 256) return;
  int n = tid >> 8;            // node
  int r = tid & 255;
  int t = r >> 5;              // time step
  int c = r & 31;              // channel
  XT[tid] = X[(size_t)t * Nn * 32 + (size_t)n * 32 + c];
}

// ---------------- 2. degree count ------------------------------------------
__global__ __launch_bounds__(256) void count_deg(
    const int* __restrict__ dst, float* __restrict__ cnt, int E) {
  int e = blockIdx.x * blockDim.x + threadIdx.x;
  if (e >= E) return;
  atomicAdd(cnt + dst[e], 1.0f);
}

// ---------------- 3/6. edge-weighted scatter aggregation -------------------
// One wave32 per edge: lane i handles features i, i+32, ... (coalesced rows,
// rows are F*4 bytes and contiguous -> global_load_b32 + global_atomic_add_f32,
// all L2-resident on the 192MB L2).
template <int F>
__global__ __launch_bounds__(256) void edge_agg(
    const int* __restrict__ src, const int* __restrict__ dst,
    const float* __restrict__ ew, const float* __restrict__ xin,
    float* __restrict__ agg, int E) {
  int wid  = (blockIdx.x * blockDim.x + threadIdx.x) >> 5;
  int lane = threadIdx.x & 31;
  if (wid >= E) return;
  int   s = src[wid];
  int   d = dst[wid];
  float w = ew[wid];
  const float* xs = xin + (size_t)s * F;
  float*       ad = agg + (size_t)d * F;
  // gfx1250 speculative prefetch of a future edge's source row
  int pe = wid + 64;
  if (pe < E && lane == 0)
    __builtin_prefetch(xin + (size_t)src[pe] * F, 0, 1);
#pragma unroll
  for (int i = lane; i < F; i += 32)
    atomicAdd(ad + i, w * xs[i]);
}

// ---------------- 4. pack h1 = concat(x, agg/deg) as f16 -------------------
__global__ __launch_bounds__(256) void pack_h1(
    const float* __restrict__ XT, const float* __restrict__ agg1,
    const float* __restrict__ cnt, _Float16* __restrict__ h1, int Nn) {
  int idx = blockIdx.x * blockDim.x + threadIdx.x;
  if (idx >= Nn * 512) return;
  int n = idx >> 9;
  int r = idx & 511;
  int t = r >> 6;
  int j = r & 63;
  float inv = 1.0f / fmaxf(cnt[n], 1.0f);
  float v = (j < 32) ? XT[(size_t)n * 256 + t * 32 + j]
                     : agg1[(size_t)n * 256 + t * 32 + (j - 32)] * inv;
  h1[idx] = (_Float16)v;
}

// ---------------- 7. pack agg2/deg into h2 cols 16..31 ---------------------
__global__ __launch_bounds__(256) void pack_h2(
    const float* __restrict__ agg2, const float* __restrict__ cnt,
    _Float16* __restrict__ h2, int M) {
  int idx = blockIdx.x * blockDim.x + threadIdx.x;
  if (idx >= M * 16) return;
  int m = idx >> 4;
  int j = idx & 15;
  int n = m >> 3;  // m = n*T + t, T = 8
  h2[(size_t)m * 32 + 16 + j] = (_Float16)(agg2[idx] / fmaxf(cnt[n], 1.0f));
}

// ------------- per-lane fragment helpers (ISA 7.12.2 layouts) --------------
// A(16x32,f16): lane L -> row L&15; kbase = (L<16)?0:8; element i holds
//   K = kbase + i (i<8) or kbase + 8 + i (i>=8)   -> two contiguous 16B loads.
__device__ inline v16h load_a_frag(const _Float16* arow /* = &A[row][kbase] */) {
  h8 lo = *(const h8*)(arow);        // K = kbase .. kbase+7
  h8 hi = *(const h8*)(arow + 16);   // K = kbase+16 .. kbase+23
  v16h a;
#pragma unroll
  for (int i = 0; i < 8; ++i) { a[i] = lo[i]; a[8 + i] = hi[i]; }
  return a;
}

// ---------------- 5. GEMM1: [M x 64] x [64 x 16] + bias + leaky ------------
// One wave per 16-row tile; K=64 -> two v_wmma_f32_16x16x32_f16 accumulating
// into the same v8f C. Epilogue writes y1 (f32) and h2[:,0:16] (f16).
__global__ __launch_bounds__(256) void gemm1_wmma(
    const _Float16* __restrict__ h1, const float* __restrict__ W1,
    const float* __restrict__ b1, float* __restrict__ y1,
    _Float16* __restrict__ h2, int Mtiles) {
  int wave = (blockIdx.x * blockDim.x + threadIdx.x) >> 5;
  if (wave >= Mtiles) return;
  int lane   = threadIdx.x & 31;
  int m0     = wave * 16;
  int row    = lane & 15;
  int col    = lane & 15;
  int kbaseA = (lane < 16) ? 0 : 8;
  int kbaseB = (lane < 16) ? 0 : 8;   // B(32x16): same K order as A (kbase+i, kbase+16+i-8)

  v8f c = {};
#pragma unroll
  for (int s = 0; s < 2; ++s) {
    v16h a = load_a_frag(h1 + (size_t)(m0 + row) * 64 + 32 * s + kbaseA);
    v16h b;
#pragma unroll
    for (int i = 0; i < 16; ++i)
      b[i] = (_Float16)W1[(32 * s + kbaseB + i + (i >= 8 ? 8 : 0)) * 16 + col];
    c = __builtin_amdgcn_wmma_f32_16x16x32_f16(
        /*neg_a=*/false, a, /*neg_b=*/false, b,
        /*c_mod=*/(short)0, c, /*reuse_a=*/false, /*reuse_b=*/false);
  }
  float bias  = b1[col];
  int   rbase = (lane < 16) ? 0 : 8;  // C/D: vgpr r -> M = r (+8 for hi lanes)
#pragma unroll
  for (int r = 0; r < 8; ++r) {
    int   m = m0 + rbase + r;
    float v = c[r] + bias;
    v = LEAKY(v);
    y1[(size_t)m * 16 + col] = v;             // fp32 for layer-2 aggregation
    h2[(size_t)m * 32 + col] = (_Float16)v;   // f16 A-operand for GEMM2
  }
}

// ---------------- 8. GEMM2: [M x 32] x [32 x 64] + bias + leaky + transpose
// One wave per 16-row tile; single A fragment reused across 4 B tiles
// (4 WMMAs). Writes straight to output [T, N, 64].
__global__ __launch_bounds__(256) void gemm2_wmma(
    const _Float16* __restrict__ h2, const float* __restrict__ W2,
    const float* __restrict__ b2, float* __restrict__ out,
    int Mtiles, int Nn) {
  int wave = (blockIdx.x * blockDim.x + threadIdx.x) >> 5;
  if (wave >= Mtiles) return;
  int lane   = threadIdx.x & 31;
  int m0     = wave * 16;
  int row    = lane & 15;
  int col    = lane & 15;
  int kbaseA = (lane < 16) ? 0 : 8;
  int kbaseB = (lane < 16) ? 0 : 8;
  int rbase  = (lane < 16) ? 0 : 8;

  v16h a = load_a_frag(h2 + (size_t)(m0 + row) * 32 + kbaseA);

#pragma unroll
  for (int j = 0; j < 4; ++j) {  // 4 column tiles of 16 -> 64 outputs
    v16h b;
#pragma unroll
    for (int i = 0; i < 16; ++i)
      b[i] = (_Float16)W2[(kbaseB + i + (i >= 8 ? 8 : 0)) * 64 + 16 * j + col];
    v8f c = {};
    c = __builtin_amdgcn_wmma_f32_16x16x32_f16(
        false, a, false, b, (short)0, c, false, false);
    float bias = b2[16 * j + col];
#pragma unroll
    for (int r = 0; r < 8; ++r) {
      int   m = m0 + rbase + r;
      int   n = m >> 3;   // m = n*8 + t
      int   t = m & 7;
      float v = c[r] + bias;
      v = LEAKY(v);
      out[(size_t)t * Nn * 64 + (size_t)n * 64 + 16 * j + col] = v;
    }
  }
}

// ---------------------------------------------------------------------------
extern "C" void kernel_launch(void* const* d_in, const int* in_sizes, int n_in,
                              void* d_out, int out_size, void* d_ws, size_t ws_size,
                              hipStream_t stream) {
  const float* X  = (const float*)d_in[0];
  const int*   EI = (const int*)d_in[1];   // [2, E]: row0 = src, row1 = dst
  const float* EW = (const float*)d_in[2];
  const float* W1 = (const float*)d_in[3];
  const float* B1 = (const float*)d_in[4];
  const float* W2 = (const float*)d_in[5];
  const float* B2 = (const float*)d_in[6];
  float* out = (float*)d_out;

  const int E  = in_sizes[1] / 2;
  const int T  = 8;
  const int Nn = in_sizes[0] / (T * 32);   // 50000
  const int M  = Nn * T;                   // 400000 (multiple of 16)
  const int Mtiles = M / 16;               // 25000

  const int* src = EI;
  const int* dst = EI + E;

  // ---- workspace layout (bytes) ----
  // cnt: N*4 | region A: N*1024 (XT, later y1 + h2) | region B: N*1024 (agg1,
  // later agg2) | region C: N*1024 (h1 f16).  Total ~= 154 MB.
  char*  ws    = (char*)d_ws;
  size_t offA  = ((size_t)Nn * 4 + 255) & ~(size_t)255;
  size_t offB  = offA + (size_t)Nn * 1024;
  size_t offC  = offB + (size_t)Nn * 1024;
  float*     cnt  = (float*)ws;
  float*     XT   = (float*)(ws + offA);               // [N, 256] f32
  float*     y1   = (float*)(ws + offA);               // [M, 16] f32 (reuses XT)
  _Float16*  h2   = (_Float16*)(ws + offA + (size_t)Nn * 512);  // [M, 32] f16
  float*     agg1 = (float*)(ws + offB);               // [N, 256] f32
  float*     agg2 = (float*)(ws + offB);               // [M, 16] f32 (reuse)
  _Float16*  h1   = (_Float16*)(ws + offC);            // [M, 64] f16

  const int TB = 256;

  // 0) zero accumulators
  hipMemsetAsync(cnt,  0, (size_t)Nn * 4,    stream);
  hipMemsetAsync(agg1, 0, (size_t)Nn * 1024, stream);

  // 1) transpose X -> XT [N, T*C]
  transpose_x<<<(Nn * 256 + TB - 1) / TB, TB, 0, stream>>>(X, XT, Nn);

  // 2) degrees
  count_deg<<<(E + TB - 1) / TB, TB, 0, stream>>>(dst, cnt, E);

  // 3) layer-1 aggregation (wave per edge, 256 feats)
  edge_agg<256><<<(E * 32 + TB - 1) / TB, TB, 0, stream>>>(src, dst, EW, XT, agg1, E);

  // 4) pack h1 (f16) = [x | agg/deg]
  pack_h1<<<(Nn * 512 + TB - 1) / TB, TB, 0, stream>>>(XT, agg1, cnt, h1, Nn);

  // 5) GEMM1 (WMMA) -> y1 (f32) + h2[:,0:16] (f16); overwrites XT region (done with it)
  gemm1_wmma<<<(Mtiles * 32 + TB - 1) / TB, TB, 0, stream>>>(h1, W1, B1, y1, h2, Mtiles);

  // 6) layer-2 aggregation over y1 (wave per edge, 128 feats); agg2 reuses agg1
  hipMemsetAsync(agg2, 0, (size_t)Nn * 512, stream);
  edge_agg<128><<<(E * 32 + TB - 1) / TB, TB, 0, stream>>>(src, dst, EW, y1, agg2, E);

  // 7) pack h2[:,16:32]
  pack_h2<<<(M * 16 + TB - 1) / TB, TB, 0, stream>>>(agg2, cnt, h2, M);

  // 8) GEMM2 (WMMA) -> out [T, N, 64] with fused bias/leaky/transpose
  gemm2_wmma<<<(Mtiles * 32 + TB - 1) / TB, TB, 0, stream>>>(h2, W2, B2, out, Mtiles, Nn);
}
